// critic_attention_new_49409303773228
// MI455X (gfx1250) — matrix operand order repaired and checked by reading, not compiled, not run on hardware
//
#include <hip/hip_runtime.h>
#include <hip/hip_bf16.h>

// ---------------------------------------------------------------------------
// MI455X (gfx1250) fused critic-attention kernel.
// Memory-bound problem (~168MB streamed vs ~10 GFLOP): single fused kernel,
// f16 WMMA for all GEMMs, non-temporal input streaming, weights staged in LDS
// as WMMA B-fragments, attention done in D-fragment layout (online softmax).
// ---------------------------------------------------------------------------

typedef _Float16 h8_t  __attribute__((ext_vector_type(8)));
typedef _Float16 h16_t __attribute__((ext_vector_type(16)));
typedef float    f8_t  __attribute__((ext_vector_type(8)));
typedef float    f4_t  __attribute__((ext_vector_type(4)));

#define B_TOT    65536
#define N_AG     16
#define OD       32
#define AD       8
#define H_DIM    128
#define HH_DIM   64
#define NTHREADS 256
#define NWAVES   8
#define SLOPE    0.01f

static __device__ __forceinline__ f8_t splat8(float v){
  f8_t r;
#pragma unroll
  for(int i=0;i<8;++i) r[i]=v;
  return r;
}
static __device__ __forceinline__ f8_t lrelu8(f8_t x){
#pragma unroll
  for(int i=0;i<8;++i) x[i] = fmaxf(x[i],0.f) + SLOPE*fminf(x[i],0.f);
  return x;
}
static __device__ __forceinline__ f8_t tanh8(f8_t x){
#pragma unroll
  for(int i=0;i<8;++i){
#if __has_builtin(__builtin_amdgcn_tanhf)
    x[i] = __builtin_amdgcn_tanhf(x[i]);     // v_tanh_f32 (gfx1250 trans op)
#else
    x[i] = tanhf(x[i]);
#endif
  }
  return x;
}
static __device__ __forceinline__ f8_t exp8(f8_t x){
#pragma unroll
  for(int i=0;i<8;++i) x[i] = __expf(x[i]);  // v_exp_f32
  return x;
}
static __device__ __forceinline__ f8_t max8(f8_t a, f8_t b){
#pragma unroll
  for(int i=0;i<8;++i) a[i] = fmaxf(a[i],b[i]);
  return a;
}
// Sum across the 16 lanes of each half-wave group (xor bits 0..3 stay in-group).
// After this every lane holds, per vector element r, the value for batch row
// m = r + 8*(lane/16) -- exactly the broadcast shape of the D-fragment rows.
static __device__ __forceinline__ f8_t red16(f8_t v){
#pragma unroll
  for(int i=0;i<8;++i){
    float x = v[i];
    x += __shfl_xor(x,1);
    x += __shfl_xor(x,2);
    x += __shfl_xor(x,4);
    x += __shfl_xor(x,8);
    v[i]=x;
  }
  return v;
}
static __device__ __forceinline__ h8_t cvt44(f4_t a, f4_t b){
  h8_t r;
#pragma unroll
  for(int i=0;i<4;++i){ r[i]=(_Float16)a[i]; r[i+4]=(_Float16)b[i]; }
  return r;
}
static __device__ __forceinline__ h8_t zero8h(){
  h8_t r;
#pragma unroll
  for(int i=0;i<8;++i) r[i]=(_Float16)0.f;
  return r;
}
static __device__ __forceinline__ h16_t cat8(h8_t a, h8_t b){
  return __builtin_shufflevector(a,b,0,1,2,3,4,5,6,7,8,9,10,11,12,13,14,15);
}
static __device__ __forceinline__ f4_t ntl4(const float* p){
  return __builtin_nontemporal_load((const f4_t*)p);   // TH=NT: don't evict weights from L2
}
// B fragment staged frag-linear in LDS: frag f, lane -> 16 halves at f*512+lane*16.
static __device__ __forceinline__ h16_t loadB(const _Float16* base, int f, int lane){
  const h8_t* p = (const h8_t*)(base + f*512 + lane*16);
  return cat8(p[0], p[1]);                              // 2x ds_load_b128
}
// A fragment from per-wave row-major f16 scratch [16][128].
// Lane holds row m=lane%16; K pattern {b..b+7, b+16..b+23}, b = 32*s + 8*(lane/16).
static __device__ __forceinline__ h16_t loadA(const _Float16* xb, int s, int mrow, int hs){
  const int k0 = 32*s + 8*hs;
  const h8_t* p0 = (const h8_t*)(xb + mrow*H_DIM + k0);
  const h8_t* p1 = (const h8_t*)(xb + mrow*H_DIM + k0 + 16);
  return cat8(p0[0], p1[0]);
}
#define WMMA(a,b,c) __builtin_amdgcn_wmma_f32_16x16x32_f16(false,(a),false,(b),(short)0,(c),false,false)

__global__ __launch_bounds__(NTHREADS)
void critic_attn_fused_kernel(
    const float* __restrict__ obs, const float* __restrict__ act,
    const float* __restrict__ W0, const float* __restrict__ b0,
    const float* __restrict__ W1, const float* __restrict__ b1,
    const float* __restrict__ W2, const float* __restrict__ b2,
    const float* __restrict__ W3, const float* __restrict__ b3,
    const float* __restrict__ Wc, const float* __restrict__ bc,
    float* __restrict__ out)
{
  // LDS: weight fragments (f16) + per-wave transpose scratch. ~104KB/block.
  __shared__ __attribute__((aligned(16))) _Float16 sW0[2*4*512];        // 8KB
  __shared__ __attribute__((aligned(16))) _Float16 sW2[4*8*512];        // 32KB
  __shared__ __attribute__((aligned(16))) _Float16 sW3[4*8*512];        // 32KB
  __shared__ __attribute__((aligned(16))) _Float16 sX [NWAVES*16*H_DIM];// 32KB

  const int tid = threadIdx.x;

  // ---- stage weights into LDS in B-fragment layout (f32 -> f16) ----
  {
    const float* Wsrc[3]  = { W0, W2, W3 };
    _Float16*    Wdst[3]  = { sW0, sW2, sW3 };
    const int    fin[3]   = { OD+AD, H_DIM, H_DIM };
    const int    fout[3]  = { HH_DIM, H_DIM, H_DIM };
    const int    Tn[3]    = { 4, 8, 8 };
    const int    tot[3]   = { 2*4*512, 4*8*512, 4*8*512 };
#pragma unroll
    for(int j=0;j<3;++j){
      for(int i=tid;i<tot[j];i+=NTHREADS){
        int f = i>>9, rem = i&511, ln = rem>>4, e = rem&15;
        int s = f / Tn[j], t = f % Tn[j];
        int K = 32*s + 8*(ln>>4) + ((e<8)?e:8+e);
        int col = 16*t + (ln&15);
        float v = (K<fin[j]) ? Wsrc[j][K*fout[j]+col] : 0.f;
        Wdst[j][i] = (_Float16)v;
      }
    }
  }
  __syncthreads();

  const int  wv   = tid>>5;
  const int  lane = tid&31;
  const int  mrow = lane&15;     // A row / B col / D col within tile
  const int  hs   = lane>>4;     // lane half select
  const int  rb   = 8*hs;        // D-fragment row base for this lane half
  const long tile = (long)blockIdx.x*NWAVES + wv;   // 16-row batch tile
  const long row  = tile*16 + mrow;
  const float* obsr = obs + row*(N_AG*OD);
  const float* actr = act + row*(N_AG*AD);
  _Float16* xlds = sX + wv*(16*H_DIM);              // wave-private scratch

  float b0c[4], w1s[4], w1o[4];
#pragma unroll
  for(int t=0;t<4;++t){
    b0c[t]=b0[16*t+mrow];
    w1s[t]=W1[16*t+mrow];           // W1[:64]  (self half)
    w1o[t]=W1[HH_DIM+16*t+mrow];    // W1[64:]  (others half)
  }
  const float b1v=b1[0], bcv=bc[0];

  // Layer-0 weight fragments resident in VGPRs (reused for all 16 agents).
  h16_t w0f[8];
#pragma unroll
  for(int f=0;f<8;++f) w0f[f]=loadB(sW0,f,lane);

  // Per-agent transform: h[16x64] = concat(obs,act)[16x40] @ W0 + b0.
  // k-step 0 = 32 obs features (exact), k-step 1 = 8 act features zero-padded.
  auto layer0 = [&](int n, f8_t (&hh)[4]){
    const float* po = obsr + n*OD + 8*hs;
    h16_t a0 = cat8(cvt44(ntl4(po),    ntl4(po+4)),
                    cvt44(ntl4(po+16), ntl4(po+20)));
    h16_t a1;
    if(hs==0){
      const float* pa = actr + n*AD;
      a1 = cat8(cvt44(ntl4(pa), ntl4(pa+4)), zero8h());  // K=32..39 act, rest 0
    } else {
      a1 = cat8(zero8h(), zero8h());                     // K=40..63 padding
    }
#pragma unroll
    for(int t=0;t<4;++t){
      f8_t acc = splat8(b0c[t]);
      acc = WMMA(a0, w0f[t],   acc);
      acc = WMMA(a1, w0f[4+t], acc);
      hh[t]=acc;
    }
  };

  // ---- agent 0: x_self = lrelu(h0), a_self = x_self @ W1[:64] + b1 ----
  f8_t h0[4]; layer0(0,h0);
  f8_t xs[4];
#pragma unroll
  for(int t=0;t<4;++t) xs[t]=lrelu8(h0[t]);
  f8_t ps = xs[0]*w1s[0] + xs[1]*w1s[1] + xs[2]*w1s[2] + xs[3]*w1s[3];
  f8_t asel = red16(ps) + b1v;

  // ---- agents 1..15: tanh + online softmax over attention logits ----
  f8_t mr = splat8(-1e30f), lr = splat8(0.f);
  f8_t Ss[4];
#pragma unroll
  for(int t=0;t<4;++t) Ss[t]=splat8(0.f);

#pragma unroll 1
  for(int n=1;n<N_AG;++n){
    f8_t hn[4]; layer0(n,hn);
    f8_t xo0=tanh8(hn[0]), xo1=tanh8(hn[1]), xo2=tanh8(hn[2]), xo3=tanh8(hn[3]);
    f8_t p = xo0*w1o[0] + xo1*w1o[1] + xo2*w1o[2] + xo3*w1o[3];
    p = red16(p);
    f8_t lg  = lrelu8(asel + p);
    f8_t mn  = max8(mr,lg);
    f8_t scl = exp8(mr-mn);
    f8_t pe  = exp8(lg-mn);
    lr = lr*scl + pe;
    Ss[0]=Ss[0]*scl+pe*xo0; Ss[1]=Ss[1]*scl+pe*xo1;
    Ss[2]=Ss[2]*scl+pe*xo2; Ss[3]=Ss[3]*scl+pe*xo3;
    mr = mn;
  }
  f8_t inv;
#pragma unroll
  for(int i=0;i<8;++i) inv[i]=1.f/lr[i];

  // ---- x_cat = [x_self | x_sum] -> per-wave LDS (D-layout -> row-major f16) ----
#pragma unroll
  for(int t=0;t<4;++t){
    f8_t xq = Ss[t]*inv;
#pragma unroll
    for(int r=0;r<8;++r){
      xlds[(r+rb)*H_DIM + 16*t + mrow]          = (_Float16)xs[t][r];
      xlds[(r+rb)*H_DIM + HH_DIM + 16*t + mrow] = (_Float16)xq[r];
    }
  }
  asm volatile("s_wait_dscnt 0" ::: "memory");

  float b2c[8], b3c[8], wcc[8];
#pragma unroll
  for(int t=0;t<8;++t){
    b2c[t]=b2[16*t+mrow]; b3c[t]=b3[16*t+mrow]; wcc[t]=Wc[16*t+mrow];
  }

  // ---- layer 2: x1 = lrelu(x_cat @ W2 + b2), written back to scratch ----
  h16_t xa0=loadA(xlds,0,mrow,hs), xa1=loadA(xlds,1,mrow,hs),
        xa2=loadA(xlds,2,mrow,hs), xa3=loadA(xlds,3,mrow,hs);
#pragma unroll
  for(int t=0;t<8;++t){
    f8_t acc = splat8(b2c[t]);
    acc = WMMA(xa0, loadB(sW2,     t,lane), acc);
    acc = WMMA(xa1, loadB(sW2,  8+t,lane), acc);
    acc = WMMA(xa2, loadB(sW2, 16+t,lane), acc);
    acc = WMMA(xa3, loadB(sW2, 24+t,lane), acc);
    acc = lrelu8(acc);
#pragma unroll
    for(int r=0;r<8;++r) xlds[(r+rb)*H_DIM + 16*t + mrow] = (_Float16)acc[r];
  }
  asm volatile("s_wait_dscnt 0" ::: "memory");

  // ---- layer 3 + final dot: value = lrelu(x1 @ W3 + b3) @ Wc + bc ----
  h16_t ya0=loadA(xlds,0,mrow,hs), ya1=loadA(xlds,1,mrow,hs),
        ya2=loadA(xlds,2,mrow,hs), ya3=loadA(xlds,3,mrow,hs);
  f8_t part = splat8(0.f);
#pragma unroll
  for(int t=0;t<8;++t){
    f8_t acc = splat8(b3c[t]);
    acc = WMMA(ya0, loadB(sW3,     t,lane), acc);
    acc = WMMA(ya1, loadB(sW3,  8+t,lane), acc);
    acc = WMMA(ya2, loadB(sW3, 16+t,lane), acc);
    acc = WMMA(ya3, loadB(sW3, 24+t,lane), acc);
    acc = lrelu8(acc);
    part += acc * wcc[t];
  }
  part = red16(part);

  if(mrow<8){
    float v = part[0];
#pragma unroll
    for(int e=1;e<8;++e) if(mrow==e) v = part[e];
    out[tile*16 + rb + mrow] = v + bcv;
  }
}

extern "C" void kernel_launch(void* const* d_in, const int* in_sizes, int n_in,
                              void* d_out, int out_size, void* d_ws, size_t ws_size,
                              hipStream_t stream) {
  (void)in_sizes; (void)n_in; (void)out_size; (void)d_ws; (void)ws_size;
  const float* obs = (const float*)d_in[0];
  const float* act = (const float*)d_in[1];
  const float* W0  = (const float*)d_in[2];
  const float* b0  = (const float*)d_in[3];
  const float* W1  = (const float*)d_in[4];
  const float* b1  = (const float*)d_in[5];
  const float* W2  = (const float*)d_in[6];
  const float* b2  = (const float*)d_in[7];
  const float* W3  = (const float*)d_in[8];
  const float* b3  = (const float*)d_in[9];
  const float* Wc  = (const float*)d_in[10];
  const float* bc  = (const float*)d_in[11];
  float* out = (float*)d_out;

  dim3 grid(B_TOT/(16*NWAVES));   // 512 blocks x 8 waves = 4096 batch tiles
  dim3 block(NTHREADS);
  critic_attn_fused_kernel<<<grid, block, 0, stream>>>(
      obs, act, W0, b0, W1, b1, W2, b2, W3, b3, Wc, bc, out);
}
